// MultiHead_Self_Attention_55542517072056
// MI455X (gfx1250) — compile-verified, matrix operand order checked by reading, not run on hardware
//
#include <hip/hip_runtime.h>
#include <hip/hip_bf16.h>

// ---------------------------------------------------------------------------
// MHA with RoPE for MI455X (gfx1250, wave32, WMMA).
// Pipeline:
//   0) fp32 -> bf16 convert of x and Wq/Wk/Wv/Wo
//   1) fused QKV projection (bf16 WMMA, f32 accum) + RoPE
//        -> Q,K [B,H,S,64] bf16 ;  V stored TRANSPOSED [B,H,64,S] bf16 so the
//        attention P.V B-fragments become contiguous row-major b128 loads.
//   2) flash attention, one wave per (b,h,16-query-rows), 32-key blocks,
//      scores + P.V both via v_wmma_f32_16x16x32_bf16, online softmax
//   3) output projection O @ Wo^T (one wave = 16x64 tile, 4x A reuse) -> fp32
// Workspace: 48 MiB bf16 scratch.
// ---------------------------------------------------------------------------

#define D_MODEL 1024
#define N_HEADS 16
#define DKH     64
#define BATCH   2
#define SEQ     2048
#define ROWS    (BATCH * SEQ)   // 4096

typedef __attribute__((ext_vector_type(16))) __bf16 v16bf;
typedef __attribute__((ext_vector_type(8)))  float  v8f;

union FragAB {
    unsigned int u[8];
    v16bf v;
};

__device__ __forceinline__ unsigned short f32_to_bf16_rne(float f) {
    unsigned int u = __float_as_uint(f);
    unsigned int r = 0x7FFFu + ((u >> 16) & 1u);
    u += r;
    return (unsigned short)(u >> 16);
}

// K-pair base offset for 16-bit A/B fragments (ISA 7.12.2, 16-bit A 16x32):
// dword v of lane (half = lane>>4) holds elements K = kb, kb+1.
__device__ __forceinline__ int kpair_base(int v, int half) {
    return ((v & 3) << 1) + ((v >> 2) << 4) + (half << 3);
}

// Load a 16(row) x 32(k) bf16 tile, row-major in memory, into an A/B fragment.
// rowptr must already point at (this lane's row, k0). Lowers to 2x b128.
__device__ __forceinline__ void load_tile16x32(FragAB& f, const unsigned short* rowptr, int half) {
#pragma unroll
    for (int v = 0; v < 8; ++v) {
        f.u[v] = *(const unsigned int*)(rowptr + kpair_base(v, half));
    }
}

__device__ __forceinline__ v8f wmma_bf16(const FragAB& a, const FragAB& b, v8f c) {
    return __builtin_amdgcn_wmma_f32_16x16x32_bf16(
        /*neg_a=*/false, a.v, /*neg_b=*/false, b.v,
        /*c_mod=*/(short)0, c, /*reuse_a=*/false, /*reuse_b=*/false);
}

// ---------------------------------------------------------------------------
// Stage 0: fp32 -> bf16
// ---------------------------------------------------------------------------
__global__ void to_bf16_kernel(const float* __restrict__ in,
                               unsigned short* __restrict__ out, int n) {
    int i = blockIdx.x * blockDim.x + threadIdx.x;
    if (i < n) out[i] = f32_to_bf16_rne(in[i]);
}

// ---------------------------------------------------------------------------
// Stage 1: fused QKV GEMM + RoPE.
// out[b,s,o] = sum_i W[o,i] * x[b,s,i]  (A = x row-major, B = W row-major)
// One wave -> one 16x16 tile of Q, K and V (A fragment reused 3x per k-step).
// Q,K stored [B,H,S,64]; V stored transposed [B,H,64,S].
// Grid: 256 (M tiles) x 64 (N tiles) = 16384 waves, 4 waves / 128-thr block.
// ---------------------------------------------------------------------------
__global__ __launch_bounds__(128) void qkv_rope_kernel(
    const unsigned short* __restrict__ xb,
    const unsigned short* __restrict__ wqb,
    const unsigned short* __restrict__ wkb,
    const unsigned short* __restrict__ wvb,
    const int* __restrict__ tp,
    unsigned short* __restrict__ Qb,
    unsigned short* __restrict__ Kb,
    unsigned short* __restrict__ Vtb) {
    const int lane = threadIdx.x & 31;
    const int wave = threadIdx.x >> 5;
    const int tile = blockIdx.x * 4 + wave;
    const int tm = tile >> 6;   // 0..255 : 16-row tile of (b,s)
    const int tn = tile & 63;   // 0..63  : 16-col tile of o
    const int half = lane >> 4;
    const int l15 = lane & 15;

    const unsigned short* arow = xb  + (size_t)(tm * 16 + l15) * D_MODEL;
    const unsigned short* qrow = wqb + (size_t)(tn * 16 + l15) * D_MODEL;
    const unsigned short* krow = wkb + (size_t)(tn * 16 + l15) * D_MODEL;
    const unsigned short* vrow = wvb + (size_t)(tn * 16 + l15) * D_MODEL;

    v8f accQ = {}; v8f accK = {}; v8f accV = {};
    for (int k0 = 0; k0 < D_MODEL; k0 += 32) {
        FragAB a, bq, bk, bv;
        load_tile16x32(a,  arow + k0, half);
        load_tile16x32(bq, qrow + k0, half);
        load_tile16x32(bk, krow + k0, half);
        load_tile16x32(bv, vrow + k0, half);
        // stage next k-tile of the streaming rows into cache behind the WMMAs
        __builtin_prefetch(arow + k0 + 32, 0, 3);
        __builtin_prefetch(qrow + k0 + 32, 0, 3);
        accQ = wmma_bf16(a, bq, accQ);
        accK = wmma_bf16(a, bk, accK);
        accV = wmma_bf16(a, bv, accV);
    }

    // RoPE on the f32 accumulators. C layout: row = tm*16 + v + 8*half,
    // col = tn*16 + l15. The interleaved pair partner (col^1) lives in the
    // adjacent lane -> shfl_xor(…, 1).
    const int o = tn * 16 + l15;
    const int h = o >> 6;
    const int d = o & 63;
    const int p = d >> 1;
    const int parity = d & 1;
    // inv_freq = THETA^(-2p/64) = exp(-p * 2*ln(10000)/64)
    const float invf = __expf(-0.28782313662425572f * (float)p);

#pragma unroll
    for (int v = 0; v < 8; ++v) {
        const int r = tm * 16 + v + 8 * half;
        const int b = r >> 11;
        const int s = r & (SEQ - 1);
        const float pos = (float)tp[s];
        const float ang = pos * invf;
        float sn, cs;
        __sincosf(ang, &sn, &cs);

        float qv = accQ[v];
        float qp = __shfl_xor(qv, 1, 32);
        float qo = parity ? (qp * sn + qv * cs) : (qv * cs - qp * sn);

        float kv = accK[v];
        float kp = __shfl_xor(kv, 1, 32);
        float ko = parity ? (kp * sn + kv * cs) : (kv * cs - kp * sn);

        const int bh = b * N_HEADS + h;
        const size_t idx  = (((size_t)bh * SEQ) + s) * DKH + d;       // [b,h,s,d]
        const size_t idxt = (((size_t)bh * DKH) + d) * SEQ + s;       // [b,h,d,s]
        Qb[idx]   = f32_to_bf16_rne(qo);
        Kb[idx]   = f32_to_bf16_rne(ko);
        Vtb[idxt] = f32_to_bf16_rne(accV[v]);
    }
}

// ---------------------------------------------------------------------------
// Stage 2: flash attention. One wave per (b, h, 16-query-row tile).
// Key blocks of 32; scores = 4 WMMAs, P.V = 4 WMMAs per block.
// V^T layout makes every WMMA operand a contiguous row-major b128 load.
// Grid: 32 bh * 128 qtiles = 4096 waves, 4 waves / block.
// ---------------------------------------------------------------------------
__global__ __launch_bounds__(128) void attn_kernel(
    const unsigned short* __restrict__ Qb,
    const unsigned short* __restrict__ Kb,
    const unsigned short* __restrict__ Vtb,
    unsigned short* __restrict__ Ob) {
    __shared__ unsigned short plds[4][16 * 32];   // per-wave P staging (bf16)

    const int lane = threadIdx.x & 31;
    const int wave = threadIdx.x >> 5;
    const int gt = blockIdx.x * 4 + wave;
    const int bh = gt >> 7;    // 0..31
    const int qt = gt & 127;   // 0..127
    const int half = lane >> 4;
    const int l15 = lane & 15;
    const int b = bh >> 4;
    const int h = bh & 15;

    const unsigned short* Qp  = Qb  + (size_t)bh * SEQ * DKH;
    const unsigned short* Kp  = Kb  + (size_t)bh * SEQ * DKH;
    const unsigned short* Vtp = Vtb + (size_t)bh * DKH * SEQ;

    // Q fragments (16 rows x 64 d), split into two k=32 fragments; loaded once.
    FragAB aq0, aq1;
    {
        const unsigned short* qrow = Qp + (size_t)(qt * 16 + l15) * DKH;
        load_tile16x32(aq0, qrow, half);
        load_tile16x32(aq1, qrow + 32, half);
    }

    v8f oacc[4] = {{}, {}, {}, {}};   // 16 rows x 64 d output accumulators
    float mrow[8], lrow[8];
#pragma unroll
    for (int v = 0; v < 8; ++v) { mrow[v] = -__builtin_inff(); lrow[v] = 0.0f; }

    const int jmax = (qt * 16 + 15) >> 5;   // causal: last 32-key block touched
    for (int j = 0; j <= jmax; ++j) {
        const int kb0 = j * 32;

        // ---- scores: S[16q x 32k] = Q(16x64) . K(32x64)^T, two 16-key halves
        FragAB b0lo, b0hi, b1lo, b1hi;
        {
            const unsigned short* k0row = Kp + (size_t)(kb0 + l15) * DKH;
            const unsigned short* k1row = Kp + (size_t)(kb0 + 16 + l15) * DKH;
            load_tile16x32(b0lo, k0row, half);
            load_tile16x32(b0hi, k0row + 32, half);
            load_tile16x32(b1lo, k1row, half);
            load_tile16x32(b1hi, k1row + 32, half);
        }
        v8f s0 = {}, s1 = {};
        s0 = wmma_bf16(aq0, b0lo, s0);
        s0 = wmma_bf16(aq1, b0hi, s0);
        s1 = wmma_bf16(aq0, b1lo, s1);
        s1 = wmma_bf16(aq1, b1hi, s1);

        // ---- causal mask, online softmax update, stage P into LDS as bf16
        const int kc0 = kb0 + l15;
        const int kc1 = kc0 + 16;
#pragma unroll
        for (int v = 0; v < 8; ++v) {
            const int qr = qt * 16 + v + 8 * half;
            float a0 = s0[v] * 0.125f;             // D_K^-0.5 = 1/8
            float a1 = s1[v] * 0.125f;
            if (kc0 > qr) a0 = -__builtin_inff();
            if (kc1 > qr) a1 = -__builtin_inff();

            float rm = fmaxf(a0, a1);              // reduce across 16 lanes/half
            rm = fmaxf(rm, __shfl_xor(rm, 1, 32));
            rm = fmaxf(rm, __shfl_xor(rm, 2, 32));
            rm = fmaxf(rm, __shfl_xor(rm, 4, 32));
            rm = fmaxf(rm, __shfl_xor(rm, 8, 32));

            const float mnew = fmaxf(mrow[v], rm);
            const float corr = __expf(mrow[v] - mnew);
            const float p0 = __expf(a0 - mnew);
            const float p1 = __expf(a1 - mnew);
            float rs = p0 + p1;
            rs += __shfl_xor(rs, 1, 32);
            rs += __shfl_xor(rs, 2, 32);
            rs += __shfl_xor(rs, 4, 32);
            rs += __shfl_xor(rs, 8, 32);

            lrow[v] = lrow[v] * corr + rs;
            mrow[v] = mnew;
            oacc[0][v] *= corr; oacc[1][v] *= corr;
            oacc[2][v] *= corr; oacc[3][v] *= corr;

            const int prow = v + 8 * half;          // C layout -> LDS row-major
            plds[wave][prow * 32 + l15]      = f32_to_bf16_rne(p0);
            plds[wave][prow * 32 + 16 + l15] = f32_to_bf16_rne(p1);
        }
        __asm__ volatile("s_wait_dscnt 0" ::: "memory");

        // ---- reload P from LDS in A-fragment layout (16q x 32k)
        FragAB pa;
        {
            const unsigned short* pr = &plds[wave][l15 * 32];
#pragma unroll
            for (int v = 0; v < 8; ++v)
                pa.u[v] = *(const unsigned int*)(pr + kpair_base(v, half));
        }

        // ---- O += P(16x32) . V(32x64): 4 WMMAs across the d dimension.
        // B fragment wants Bt[n=d][k=key] = V^T[d][key]: contiguous in V^T.
#pragma unroll
        for (int nb = 0; nb < 4; ++nb) {
            FragAB vb;
            const unsigned short* vtrow = Vtp + (size_t)(nb * 16 + l15) * SEQ + kb0;
            load_tile16x32(vb, vtrow, half);
            oacc[nb] = wmma_bf16(pa, vb, oacc[nb]);
        }
    }

    // ---- normalize, store O as bf16 in [B,S, h*64+d] (GEMM-ready layout)
#pragma unroll
    for (int v = 0; v < 8; ++v) {
        const float inv = 1.0f / lrow[v];
        const int s = qt * 16 + v + 8 * half;
        const size_t rbase = ((size_t)(b * SEQ + s)) * D_MODEL + h * DKH;
#pragma unroll
        for (int nb = 0; nb < 4; ++nb)
            Ob[rbase + nb * 16 + l15] = f32_to_bf16_rne(oacc[nb][v] * inv);
    }
}

// ---------------------------------------------------------------------------
// Stage 3: output projection  out[b,s,o] = sum_i Wo[o,i] * O[b,s,i]
// One wave computes a 16(m) x 64(n) tile: A fragment reused across 4 WMMAs.
// Grid: 256 M-tiles x 16 N-quads = 4096 waves, 4 waves / block.
// ---------------------------------------------------------------------------
__global__ __launch_bounds__(128) void oproj_kernel(
    const unsigned short* __restrict__ Ob,
    const unsigned short* __restrict__ wob,
    float* __restrict__ out) {
    const int lane = threadIdx.x & 31;
    const int wave = threadIdx.x >> 5;
    const int tile = blockIdx.x * 4 + wave;
    const int tm = tile >> 4;     // 0..255
    const int tq = tile & 15;     // 0..15 : quad of four 16-col tiles
    const int half = lane >> 4;
    const int l15 = lane & 15;

    const unsigned short* arow = Ob + (size_t)(tm * 16 + l15) * D_MODEL;
    const unsigned short* brow[4];
#pragma unroll
    for (int nb = 0; nb < 4; ++nb)
        brow[nb] = wob + (size_t)((tq * 4 + nb) * 16 + l15) * D_MODEL;

    v8f acc[4] = {{}, {}, {}, {}};
    for (int k0 = 0; k0 < D_MODEL; k0 += 32) {
        FragAB a;
        load_tile16x32(a, arow + k0, half);
        __builtin_prefetch(arow + k0 + 32, 0, 3);
#pragma unroll
        for (int nb = 0; nb < 4; ++nb) {
            FragAB bb;
            load_tile16x32(bb, brow[nb] + k0, half);
            acc[nb] = wmma_bf16(a, bb, acc[nb]);
        }
    }

#pragma unroll
    for (int v = 0; v < 8; ++v) {
        const int r = tm * 16 + v + 8 * half;
#pragma unroll
        for (int nb = 0; nb < 4; ++nb) {
            const int o = (tq * 4 + nb) * 16 + l15;
            out[(size_t)r * D_MODEL + o] = acc[nb][v];
        }
    }
}

// ---------------------------------------------------------------------------
extern "C" void kernel_launch(void* const* d_in, const int* in_sizes, int n_in,
                              void* d_out, int out_size, void* d_ws, size_t ws_size,
                              hipStream_t stream) {
    const float* x  = (const float*)d_in[0];
    const float* Wq = (const float*)d_in[1];
    const float* Wk = (const float*)d_in[2];
    const float* Wv = (const float*)d_in[3];
    const float* Wo = (const float*)d_in[4];
    const int*   tp = (const int*)d_in[5];
    float* out = (float*)d_out;

    // Workspace layout (ushort/bf16 elements), total 48 MiB:
    unsigned short* ws  = (unsigned short*)d_ws;
    unsigned short* xb  = ws;                                          // 4096*1024
    unsigned short* wqb = xb  + (size_t)ROWS * D_MODEL;                // 1024*1024
    unsigned short* wkb = wqb + (size_t)D_MODEL * D_MODEL;
    unsigned short* wvb = wkb + (size_t)D_MODEL * D_MODEL;
    unsigned short* wob = wvb + (size_t)D_MODEL * D_MODEL;
    unsigned short* Qb  = wob + (size_t)D_MODEL * D_MODEL;             // [B,H,S,64]
    unsigned short* Kb  = Qb  + (size_t)ROWS * D_MODEL;                // [B,H,S,64]
    unsigned short* Vtb = Kb  + (size_t)ROWS * D_MODEL;                // [B,H,64,S]
    unsigned short* Obf = Vtb + (size_t)ROWS * D_MODEL;                // [B,S,1024]

    const int nx = ROWS * D_MODEL;        // 4,194,304
    const int nw = D_MODEL * D_MODEL;     // 1,048,576
    to_bf16_kernel<<<(nx + 255) / 256, 256, 0, stream>>>(x, xb, nx);
    to_bf16_kernel<<<(nw + 255) / 256, 256, 0, stream>>>(Wq, wqb, nw);
    to_bf16_kernel<<<(nw + 255) / 256, 256, 0, stream>>>(Wk, wkb, nw);
    to_bf16_kernel<<<(nw + 255) / 256, 256, 0, stream>>>(Wv, wvb, nw);
    to_bf16_kernel<<<(nw + 255) / 256, 256, 0, stream>>>(Wo, wob, nw);

    // 256 M-tiles * 64 N-tiles = 16384 waves, 4 waves per 128-thread block
    qkv_rope_kernel<<<4096, 128, 0, stream>>>(xb, wqb, wkb, wvb, tp, Qb, Kb, Vtb);

    // 32 (b,h) * 128 query tiles = 4096 waves
    attn_kernel<<<1024, 128, 0, stream>>>(Qb, Kb, Vtb, Obf);

    // 256 M-tiles * 16 N-quads = 4096 waves
    oproj_kernel<<<1024, 128, 0, stream>>>(Obf, wob, out);
}
